// MultiHeadLinearAttention_90537910599869
// MI455X (gfx1250) — matrix operand order repaired and checked by reading, not compiled, not run on hardware
//
#include <hip/hip_runtime.h>

#define EMBED   512
#define NHEADS  8
#define HDIM    64
#define SEQ     8192
#define BATCH   4
#define MTOT    (BATCH * SEQ)      // 32768 rows for the projections
#define EPSV    1e-6f
#define NBH     32                 // BATCH * NHEADS
#define LCHUNKS 8
#define LCHUNK  (SEQ / LCHUNKS)    // 1024

typedef __attribute__((ext_vector_type(16))) __bf16 v16bf;
typedef __attribute__((ext_vector_type(8)))  __bf16 v8bf;
typedef __attribute__((ext_vector_type(8)))  float  v8f;
typedef __attribute__((ext_vector_type(4)))  float  v4f;

union V16 { v16bf v; v8bf h[2]; };

// Load 8 contiguous fp32 (16B-aligned) and convert to bf16 fragment half.
__device__ inline v8bf cvt8(const float* __restrict__ p) {
    const v4f a = *(const v4f*)p;
    const v4f b = *(const v4f*)(p + 4);
    v8bf r;
    r[0] = (__bf16)a[0]; r[1] = (__bf16)a[1]; r[2] = (__bf16)a[2]; r[3] = (__bf16)a[3];
    r[4] = (__bf16)b[0]; r[5] = (__bf16)b[1]; r[6] = (__bf16)b[2]; r[7] = (__bf16)b[3];
    return r;
}

__device__ inline v8f wmma_bf16(v16bf a, v16bf b, v8f c) {
    return __builtin_amdgcn_wmma_f32_16x16x32_bf16(false, a, false, b, (short)0, c,
                                                   false, false);
}

__device__ inline v8f vzero8() {
    v8f z = {0.f, 0.f, 0.f, 0.f, 0.f, 0.f, 0.f, 0.f};
    return z;
}

// ---------------------------------------------------------------------------
// Kernel 1: phi = tanh(X @ W^T + b) + 1, output bf16.
// Workgroup tile: 128 (M) x 128 (N); 8 waves, wave w owns rows [16w, 16w+16)
// across all 8 n-tiles (8 WMMAs amortize each A-fragment).
// W staged in 128-wide k-slabs (bf16) in LDS (34 KB).
// ---------------------------------------------------------------------------
#define KSLAB 128
#define WLDST (KSLAB + 8)   // bf16 row stride: 272 B, multiple of 16

__global__ __launch_bounds__(256)
void proj_phi_kernel(const float* __restrict__ X, const float* __restrict__ W,
                     const float* __restrict__ bias, __bf16* __restrict__ phi) {
    __shared__ alignas(32) __bf16 Wl[128][WLDST];
    const int mBase = blockIdx.x * 128;
    const int nBase = blockIdx.y * 128;
    const int tid  = threadIdx.x;
    const int wave = tid >> 5;
    const int lane = tid & 31;
    const int hs   = lane >> 4;
    const int lq   = lane & 15;

    v8f acc[8];
    #pragma unroll
    for (int i = 0; i < 8; ++i) acc[i] = vzero8();

    const int rowA = mBase + wave * 16 + lq;
    const float* __restrict__ Xrow = X + (size_t)rowA * EMBED;

    for (int slab = 0; slab < EMBED; slab += KSLAB) {
        __syncthreads();
        // Stage W[nBase..+128)[slab..+128) -> bf16 LDS (coalesced over k).
        for (int r = 0; r < 64; ++r) {
            int idx = tid + 256 * r;
            int ni  = idx >> 7;        // 0..127
            int kj  = idx & 127;
            Wl[ni][kj] = (__bf16)W[(size_t)(nBase + ni) * EMBED + slab + kj];
        }
        __syncthreads();
        if (slab + KSLAB < EMBED)
            __builtin_prefetch(Xrow + slab + KSLAB, 0, 1);
        #pragma unroll
        for (int kc = 0; kc < KSLAB / 32; ++kc) {
            const int k0 = kc * 32;
            V16 a;
            a.h[0] = cvt8(Xrow + slab + k0 + 16 * hs);
            a.h[1] = cvt8(Xrow + slab + k0 + 16 * hs + 8);
            V16 bm[8];
            #pragma unroll
            for (int nt = 0; nt < 8; ++nt) {
                const __bf16* bp = &Wl[nt * 16 + lq][k0 + 16 * hs];
                bm[nt].h[0] = *(const v8bf*)bp;
                bm[nt].h[1] = *(const v8bf*)(bp + 8);
            }
            #pragma unroll
            for (int nt = 0; nt < 8; ++nt)
                acc[nt] = wmma_bf16(a.v, bm[nt].v, acc[nt]);
        }
    }
    // Epilogue: bias + tanh + 1, store bf16.
    #pragma unroll
    for (int nt = 0; nt < 8; ++nt) {
        const int n = nBase + nt * 16 + lq;
        const float bv = bias[n];
        #pragma unroll
        for (int r = 0; r < 8; ++r) {
            const int row = mBase + wave * 16 + 8 * hs + r;
            float v = acc[nt][r] + bv;
            v = tanhf(v) + 1.0f;
            phi[(size_t)row * EMBED + n] = (__bf16)v;
        }
    }
}

// ---------------------------------------------------------------------------
// Kernel 2a: partial q_global over an L-chunk: QGp[chunk][bh][d]
// ---------------------------------------------------------------------------
__global__ __launch_bounds__(256)
void qglobal_part_kernel(const float* __restrict__ Q, float* __restrict__ QGp) {
    __shared__ float red[4][64];
    const int bh = blockIdx.x, b = bh >> 3, h = bh & 7;
    const int lStart = blockIdx.y * LCHUNK;
    const int tid = threadIdx.x, d = tid & 63, seg = tid >> 6;
    const float* __restrict__ base = Q + (size_t)b * SEQ * EMBED + h * HDIM + d;
    float s = 0.f;
    for (int l = lStart + seg; l < lStart + LCHUNK; l += 4)
        s += base[(size_t)l * EMBED];
    red[seg][d] = s;
    __syncthreads();
    if (tid < 64)
        QGp[(size_t)blockIdx.y * (NBH * HDIM) + bh * HDIM + tid] =
            red[0][tid] + red[1][tid] + red[2][tid] + red[3][tid];
}

// Kernel 2b: qg[i] = (sum_c QGp[c][i]) / L / sqrt(D)   (2048 elements)
__global__ __launch_bounds__(256)
void qg_reduce_kernel(const float* __restrict__ QGp, float* __restrict__ qg) {
    const int i = blockIdx.x * 256 + threadIdx.x;
    float s = 0.f;
    #pragma unroll
    for (int c = 0; c < LCHUNKS; ++c) s += QGp[(size_t)c * (NBH * HDIM) + i];
    qg[i] = s / ((float)SEQ * 8.0f);
}

// ---------------------------------------------------------------------------
// Kernel 3a: scores[b,h,l] = qg[b,h,:] . K[b,l,h,:]  (grid 32 x 8 over L)
// ---------------------------------------------------------------------------
__global__ __launch_bounds__(256)
void scores_kernel(const float* __restrict__ K, const float* __restrict__ qg,
                   float* __restrict__ sc) {
    __shared__ float qs[HDIM];
    const int bh = blockIdx.x, b = bh >> 3, h = bh & 7;
    const int lStart = blockIdx.y * LCHUNK;
    const int tid = threadIdx.x;
    if (tid < HDIM) qs[tid] = qg[bh * HDIM + tid];
    __syncthreads();
    const float* __restrict__ Kb = K + (size_t)b * SEQ * EMBED + h * HDIM;
    for (int l = lStart + tid; l < lStart + LCHUNK; l += 256) {
        const v4f* kr = (const v4f*)(Kb + (size_t)l * EMBED);
        float s = 0.f;
        #pragma unroll
        for (int d4 = 0; d4 < 16; ++d4) {
            v4f kx = kr[d4];
            s += qs[4*d4+0]*kx[0] + qs[4*d4+1]*kx[1] + qs[4*d4+2]*kx[2] + qs[4*d4+3]*kx[3];
        }
        sc[(size_t)bh * SEQ + l] = s;
    }
}

// Kernel 3b: alpha = softmax(scores) * L, one WG per (b,h); scores held in LDS.
__global__ __launch_bounds__(256)
void alpha_fin_kernel(const float* __restrict__ sc, float* __restrict__ alpha) {
    __shared__ float scl[SEQ];
    __shared__ float red[256];
    const int bh = blockIdx.x;
    const int tid = threadIdx.x;
    const float* __restrict__ sg = sc + (size_t)bh * SEQ;
    float lmax = -3.4e38f;
    for (int l = tid; l < SEQ; l += 256) {
        float s = sg[l];
        scl[l] = s;
        lmax = fmaxf(lmax, s);
    }
    red[tid] = lmax; __syncthreads();
    for (int off = 128; off > 0; off >>= 1) {
        if (tid < off) red[tid] = fmaxf(red[tid], red[tid + off]);
        __syncthreads();
    }
    const float gmax = red[0];
    __syncthreads();
    float lsum = 0.f;
    for (int l = tid; l < SEQ; l += 256) {
        float e = __expf(scl[l] - gmax);
        scl[l] = e;
        lsum += e;
    }
    red[tid] = lsum; __syncthreads();
    for (int off = 128; off > 0; off >>= 1) {
        if (tid < off) red[tid] += red[tid + off];
        __syncthreads();
    }
    const float scale = (float)SEQ / red[0];
    for (int l = tid; l < SEQ; l += 256)
        alpha[(size_t)bh * SEQ + l] = scl[l] * scale;
}

// ---------------------------------------------------------------------------
// Kernel 4: per-(b,h) partial KV^T over an L-chunk of 1024 (grid 32 x 8),
// with fused Ksum partials. A = (phiK*alpha)^T and B = V^T staged transposed
// in LDS (contiguous in k=l), 64 l-values per stage.
// Partials: KVp[chunk][bh][e][d] (transposed: final GEMM B-fragments are then
// contiguous in k=d), KSp[chunk][bh][d].
// ---------------------------------------------------------------------------
#define KVL   64
#define KVST  (KVL + 8)   // bf16 row stride: 144 B, multiple of 16

__global__ __launch_bounds__(256)
void kv_kernel(const __bf16* __restrict__ phiK, const float* __restrict__ V,
               const float* __restrict__ alpha, float* __restrict__ KVp,
               float* __restrict__ KSp) {
    __shared__ alignas(32) __bf16 Kt[HDIM][KVST];
    __shared__ alignas(32) __bf16 Vt[HDIM][KVST];
    __shared__ float ksred[4][64];
    const int bh = blockIdx.x, b = bh >> 3, h = bh & 7;
    const int lStart = blockIdx.y * LCHUNK;
    const int tid  = threadIdx.x;
    const int wave = tid >> 5, lane = tid & 31;
    const int hs = lane >> 4, lq = lane & 15;
    const int mt = (2 * wave) >> 2;          // both tiles of a wave share mt
    const int nt0 = (2 * wave) & 3, nt1 = nt0 + 1;
    const int dStage = tid & 63;             // staging: fixed d per thread

    v8f acc0 = vzero8(), acc1 = vzero8();
    float ksPart = 0.f;
    const size_t gbase = (size_t)b * SEQ * EMBED + h * HDIM;
    const float* __restrict__ al = alpha + (size_t)bh * SEQ;

    for (int l0 = lStart; l0 < lStart + LCHUNK; l0 += KVL) {
        __syncthreads();
        #pragma unroll
        for (int r = 0; r < 16; ++r) {
            int i = (tid >> 6) + 4 * r;          // l offset 0..63
            size_t g = gbase + (size_t)(l0 + i) * EMBED + dStage;
            float kval = (float)phiK[g] * al[l0 + i];
            ksPart += kval;
            Kt[dStage][i] = (__bf16)kval;
            Vt[dStage][i] = (__bf16)V[g];
        }
        __syncthreads();
        #pragma unroll
        for (int kc = 0; kc < 2; ++kc) {
            const int k0 = kc * 32;
            V16 a, b0, b1;
            const int dA = mt * 16 + lq;
            a.h[0] = *(const v8bf*)&Kt[dA][k0 + 16 * hs];
            a.h[1] = *(const v8bf*)&Kt[dA][k0 + 16 * hs + 8];
            const int e0 = nt0 * 16 + lq;
            b0.h[0] = *(const v8bf*)&Vt[e0][k0 + 16 * hs];
            b0.h[1] = *(const v8bf*)&Vt[e0][k0 + 16 * hs + 8];
            const int e1 = nt1 * 16 + lq;
            b1.h[0] = *(const v8bf*)&Vt[e1][k0 + 16 * hs];
            b1.h[1] = *(const v8bf*)&Vt[e1][k0 + 16 * hs + 8];
            acc0 = wmma_bf16(a.v, b0.v, acc0);
            acc1 = wmma_bf16(a.v, b1.v, acc1);
        }
    }
    // Store transposed partial KV: KVp[chunk][bh][e*64 + d]
    float* __restrict__ outp = KVp + (size_t)blockIdx.y * (NBH * HDIM * HDIM)
                                   + (size_t)bh * (HDIM * HDIM);
    const int dBase = mt * 16 + 8 * hs;
    #pragma unroll
    for (int r = 0; r < 8; ++r) {
        outp[(nt0 * 16 + lq) * HDIM + dBase + r] = acc0[r];
        outp[(nt1 * 16 + lq) * HDIM + dBase + r] = acc1[r];
    }
    // Reduce fused Ksum partials (thread's d was fixed all along).
    ksred[tid >> 6][dStage] = ksPart;
    __syncthreads();
    if (tid < 64)
        KSp[(size_t)blockIdx.y * (NBH * HDIM) + bh * HDIM + tid] =
            ksred[0][tid] + ksred[1][tid] + ksred[2][tid] + ksred[3][tid];
}

__global__ __launch_bounds__(256)
void kv_reduce_kernel(const float* __restrict__ KVp, float* __restrict__ KVt) {
    const int i = blockIdx.x * 256 + threadIdx.x;   // 131072 elems
    float s = 0.f;
    #pragma unroll
    for (int c = 0; c < LCHUNKS; ++c) s += KVp[(size_t)c * (NBH * HDIM * HDIM) + i];
    KVt[i] = s;
}

__global__ __launch_bounds__(256)
void ks_reduce_kernel(const float* __restrict__ KSp, float* __restrict__ ksum) {
    const int i = blockIdx.x * 256 + threadIdx.x;   // 2048 elems
    float s = 0.f;
    #pragma unroll
    for (int c = 0; c < LCHUNKS; ++c) s += KSp[(size_t)c * (NBH * HDIM) + i];
    ksum[i] = s;
}

// ---------------------------------------------------------------------------
// Kernel 5: out[b,l,h,e] = (phiQ[b,l,h,:] @ KV[:,e]) / (phiQ . Ksum + eps)
// den is fused: each lane dots its own A-fragment (which IS phiQ) against
// Ksum (LDS); hs=0/hs=1 halves of each row combine via per-wave LDS.
// ---------------------------------------------------------------------------
__global__ __launch_bounds__(256)
void out_kernel(const __bf16* __restrict__ phiQ, const float* __restrict__ KVt,
                const float* __restrict__ ksum, float* __restrict__ out) {
    __shared__ float ks[HDIM];
    __shared__ float denw[8][2][16];
    const int bh = blockIdx.y, b = bh >> 3, h = bh & 7;
    const int lBase = blockIdx.x * 128;
    const int tid  = threadIdx.x;
    const int wave = tid >> 5, lane = tid & 31;
    const int hs = lane >> 4, lq = lane & 15;
    if (tid < HDIM) ks[tid] = ksum[bh * HDIM + tid];
    __syncthreads();

    const int rowA = lBase + wave * 16 + lq;
    const __bf16* __restrict__ qrow = phiQ + (size_t)(b * SEQ + rowA) * EMBED + h * HDIM;
    const float* __restrict__ kv = KVt + (size_t)bh * HDIM * HDIM;

    v8f acc[4] = {vzero8(), vzero8(), vzero8(), vzero8()};
    float dpart = 0.f;
    #pragma unroll
    for (int kc = 0; kc < 2; ++kc) {
        const int k0 = kc * 32;
        V16 a;
        a.h[0] = *(const v8bf*)(qrow + k0 + 8 * hs);
        a.h[1] = *(const v8bf*)(qrow + k0 + 16 + 8 * hs);
        #pragma unroll
        for (int i = 0; i < 8; ++i) {
            dpart += (float)a.h[0][i] * ks[k0 + 8 * hs + i];
            dpart += (float)a.h[1][i] * ks[k0 + 16 + 8 * hs + i];
        }
        #pragma unroll
        for (int nt = 0; nt < 4; ++nt) {
            const float* bp = kv + (nt * 16 + lq) * HDIM + k0 + 8 * hs;
            V16 bm;
            bm.h[0] = cvt8(bp);
            bm.h[1] = cvt8(bp + 16);
            acc[nt] = wmma_bf16(a.v, bm.v, acc[nt]);
        }
    }
    denw[wave][hs][lq] = dpart;
    __syncthreads();
    #pragma unroll
    for (int r = 0; r < 8; ++r) {
        const float dv = denw[wave][0][8 * hs + r] + denw[wave][1][8 * hs + r] + EPSV;
        const float inv = 1.0f / dv;
        const int orow = lBase + wave * 16 + 8 * hs + r;
        float* __restrict__ op = out + (size_t)(b * SEQ + orow) * EMBED + h * HDIM;
        #pragma unroll
        for (int nt = 0; nt < 4; ++nt)
            op[nt * 16 + lq] = acc[nt][r] * inv;
    }
}

// ---------------------------------------------------------------------------
extern "C" void kernel_launch(void* const* d_in, const int* in_sizes, int n_in,
                              void* d_out, int out_size, void* d_ws, size_t ws_size,
                              hipStream_t stream) {
    const float* Q  = (const float*)d_in[0];
    const float* K  = (const float*)d_in[1];
    const float* V  = (const float*)d_in[2];
    const float* Wq = (const float*)d_in[3];
    const float* bq = (const float*)d_in[4];
    const float* Wk = (const float*)d_in[5];
    const float* bk = (const float*)d_in[6];
    float* out = (float*)d_out;

    char* ws = (char*)d_ws;
    // Workspace layout (bytes):
    __bf16* phiQ = (__bf16*)(ws);                     // 32 MB
    __bf16* phiK = (__bf16*)(ws + 33554432);          // 32 MB
    float*  qg   = (float*)(ws + 67108864);           // 2048 * 4
    float*  alph = (float*)(ws + 67117056);           // 262144 * 4
    float*  ksum = (float*)(ws + 68165632);           // 2048 * 4
    float*  KVt  = (float*)(ws + 68173824);           // 131072 * 4
    float*  sc   = (float*)(ws + 68698112);           // 262144 * 4
    float*  KVp  = (float*)(ws + 69746688);           // 8 * 131072 * 4 = 4 MB
    float*  QGp  = (float*)(ws + 73940992);           // 8 * 2048 * 4 = 64 KB
    float*  KSp  = (float*)(ws + 74006528);           // 8 * 2048 * 4 = 64 KB

    const dim3 gProj(MTOT / 128, EMBED / 128);
    proj_phi_kernel<<<gProj, 256, 0, stream>>>(Q, Wq, bq, phiQ);
    proj_phi_kernel<<<gProj, 256, 0, stream>>>(K, Wk, bk, phiK);
    qglobal_part_kernel<<<dim3(NBH, LCHUNKS), 256, 0, stream>>>(Q, QGp);
    qg_reduce_kernel<<<8, 256, 0, stream>>>(QGp, qg);
    scores_kernel<<<dim3(NBH, LCHUNKS), 256, 0, stream>>>(K, qg, sc);
    alpha_fin_kernel<<<NBH, 256, 0, stream>>>(sc, alph);
    kv_kernel<<<dim3(NBH, LCHUNKS), 256, 0, stream>>>(phiK, V, alph, KVp, KSp);
    kv_reduce_kernel<<<512, 256, 0, stream>>>(KVp, KVt);
    ks_reduce_kernel<<<8, 256, 0, stream>>>(KSp, ksum);
    out_kernel<<<dim3(SEQ / 128, NBH), 256, 0, stream>>>(phiQ, KVt, ksum, out);
}
